// TransformerProjector_4887672783362
// MI455X (gfx1250) — compile-verified, matrix operand order checked by reading, not run on hardware
//
#include <hip/hip_runtime.h>
#include <math.h>
#include <stdint.h>

// ---------------------------------------------------------------------------
// Types for CDNA5 WMMA (wave32, 16x16x32 bf16 -> f32) and TDM descriptors
// ---------------------------------------------------------------------------
typedef __attribute__((ext_vector_type(16))) __bf16 v16bf;
typedef __attribute__((ext_vector_type(8)))  float  v8f;
typedef unsigned int v4u __attribute__((ext_vector_type(4)));
typedef int          v8i __attribute__((ext_vector_type(8)));
typedef int          v4i __attribute__((ext_vector_type(4)));

union Frag {
    uint4 u[2];   // 32 bytes
    v16bf v;
};

__device__ inline v8f wmma_bf16(const Frag& a, const Frag& b, v8f c) {
    return __builtin_amdgcn_wmma_f32_16x16x32_bf16(
        /*neg_a=*/false, a.v, /*neg_b=*/false, b.v,
        /*c_mod=*/(short)0, c, /*reuse_a=*/false, /*reuse_b=*/false);
}

__device__ inline unsigned short f32_to_bf16(float f) {
    union { float f; unsigned int u; } x; x.f = f;
    unsigned int r = x.u + 0x7FFFu + ((x.u >> 16) & 1u);   // round to nearest even
    return (unsigned short)(r >> 16);
}

__device__ inline float gelu_exact(float v) {
    return 0.5f * v * (1.0f + erff(v * 0.70710678118654752f));
}

// Async global->LDS (ASYNCcnt path): 16B per lane, no VGPR round-trip.
__device__ inline void async_load_lds_b128(unsigned lds_off, const void* gptr) {
    asm volatile("global_load_async_to_lds_b128 %0, %1, off"
                 :: "v"(lds_off), "v"((unsigned long long)(uintptr_t)gptr)
                 : "memory");
}
__device__ inline void wait_asynccnt0() {
    asm volatile("s_wait_asynccnt 0x0" ::: "memory");
}

// Problem constants
#define TB   4          // batch
#define TS   896        // seq
#define TT   1024       // tokens per batch = M + S
#define TD   1024       // hidden
#define TH   16         // heads
#define TDH  64         // head dim
#define TI   4096       // ffn inner
#define TOUT 4096       // projector out
#define TM   128        // memory tokens
#define NTOK 4096       // TB*TT

// ---------------------------------------------------------------------------
// Elementwise helpers
// ---------------------------------------------------------------------------
__global__ void cvt_bf16_kernel(const float* __restrict__ src,
                                unsigned short* __restrict__ dst, int n) {
    int i = blockIdx.x * blockDim.x + threadIdx.x;
    if (i < n) dst[i] = f32_to_bf16(src[i]);
}

__global__ void build_x_kernel(const float* __restrict__ hidden,
                               const float* __restrict__ mem,
                               float* __restrict__ xf,
                               unsigned short* __restrict__ xb) {
    int i = blockIdx.x * blockDim.x + threadIdx.x;   // over NTOK*TD
    int c = i & (TD - 1);
    int row = i >> 10;
    int t = row & (TT - 1);
    int b = row >> 10;
    float v = (t < TM) ? mem[t * TD + c]
                       : hidden[((size_t)b * TS + (t - TM)) * TD + c];
    xf[i] = v;
    xb[i] = f32_to_bf16(v);
}

__global__ void readout_kernel(const float* __restrict__ xf,
                               float* __restrict__ dst) {
    int i = blockIdx.x * blockDim.x + threadIdx.x;   // over TB*TM*TD
    int c = i & (TD - 1);
    int r = i >> 10;
    int t = r & (TM - 1);
    int b = r >> 7;
    dst[i] = xf[((size_t)b * TT + t) * TD + c];
}

// Fused residual + LayerNorm: x = LN(C + x) * g + b ; also emit bf16 copy
__global__ __launch_bounds__(256)
void resid_ln_kernel(const float* __restrict__ C,
                     float* __restrict__ xf,
                     unsigned short* __restrict__ xb,
                     const float* __restrict__ g,
                     const float* __restrict__ bt) {
    __shared__ float red[256];
    const int row = blockIdx.x;
    const int tid = threadIdx.x;
    const size_t base = (size_t)row * TD;

    float v[4];
    float s = 0.f;
    for (int i = 0; i < 4; ++i) {
        int c = tid + i * 256;
        v[i] = C[base + c] + xf[base + c];
        s += v[i];
    }
    red[tid] = s; __syncthreads();
    for (int off = 128; off > 0; off >>= 1) {
        if (tid < off) red[tid] += red[tid + off];
        __syncthreads();
    }
    const float mu = red[0] * (1.0f / TD);
    __syncthreads();

    float sq = 0.f;
    for (int i = 0; i < 4; ++i) { float d = v[i] - mu; sq += d * d; }
    red[tid] = sq; __syncthreads();
    for (int off = 128; off > 0; off >>= 1) {
        if (tid < off) red[tid] += red[tid + off];
        __syncthreads();
    }
    const float rs = rsqrtf(red[0] * (1.0f / TD) + 1e-12f);

    for (int i = 0; i < 4; ++i) {
        int c = tid + i * 256;
        float y = (v[i] - mu) * rs * g[c] + bt[c];
        xf[base + c] = y;
        xb[base + c] = f32_to_bf16(y);
    }
}

// ---------------------------------------------------------------------------
// bf16 GEMM:  C(MxN) = A(MxK) @ B(KxN) + bias, f32 WMMA accumulate.
// Block tile 128x128x32, 8 waves (256 threads), each wave computes 32x64
// (2x4 WMMA accumulators -> 8 v_wmma per k-step per wave).
// A tile (128x32, row-major-contiguous) is DMA'd into LDS by the Tensor Data
// Mover (one tensor_load_to_lds per k-step, wave 0 only, TENSORcnt sync).
// B tile is staged transposed (Bt[n][k]) so WMMA B-frag halves are contiguous
// 16B runs -> ds_load_b128.
// Epilogue modes: 0=f32 raw, 1=bf16 raw, 2=gelu->bf16,
//                 3=gelu->f32 with memory-token row remap (final projector)
// ---------------------------------------------------------------------------
__global__ __launch_bounds__(256)
void gemm_bf16_kernel(const unsigned short* __restrict__ A,
                      const unsigned short* __restrict__ Bm,
                      const float* __restrict__ bias,
                      void* __restrict__ out,
                      int Kdim, int Ndim, int mode) {
    __shared__ __align__(16) unsigned short As[128 * 32];   // 8 KB
    __shared__ __align__(16) unsigned short Bt[128 * 32];   // 8 KB

    const int tid  = threadIdx.x;
    const int wave = tid >> 5;
    const int lane = tid & 31;
    const int l16  = lane & 15;
    const int hi   = lane >> 4;
    const int kb8  = hi * 8;

    const int mOff = (wave >> 1) * 32;   // 0,32,64,96
    const int nOff = (wave & 1) * 64;    // 0,64
    const int blockM = blockIdx.x * 128;
    const int blockN = blockIdx.y * 128;

    v8f acc[2][4];
    for (int mi = 0; mi < 2; ++mi)
        for (int ni = 0; ni < 4; ++ni)
            for (int e = 0; e < 8; ++e) acc[mi][ni][e] = 0.f;

    const unsigned asLds = (unsigned)(uintptr_t)&As[0];

    for (int k0 = 0; k0 < Kdim; k0 += 32) {
        __syncthreads();
        // --- A tile via Tensor Data Mover (wave 0 only; TDM ignores EXEC,
        //     so the issue must be branch-guarded to a single wave) ---
        if (tid < 32) {
            const unsigned long long gaddr =
                (unsigned long long)(uintptr_t)(A + (size_t)blockM * Kdim + k0);
            v4u g0;
            g0[0] = 1u;                                   // count=1, user mode
            g0[1] = asLds;                                // lds_addr (bytes)
            g0[2] = (unsigned)(gaddr & 0xFFFFFFFFull);    // global_addr[31:0]
            g0[3] = (unsigned)((gaddr >> 32) & 0x1FFFFFFull)
                    | (2u << 30);                         // addr[56:32] | type=2
            v8i g1;
            const unsigned td0 = (unsigned)Kdim;          // tensor_dim0
            const unsigned td1 = 0x100000u;               // tensor_dim1 (OOB bound)
            g1[0] = (int)(1u << 16);                      // data_size=1 (2 bytes)
            g1[1] = (int)((td0 & 0xFFFFu) << 16);         // tensor_dim0[15:0]
            g1[2] = (int)((td0 >> 16) | ((td1 & 0xFFFFu) << 16));
            g1[3] = (int)((td1 >> 16) | (32u << 16));     // tile_dim0 = 32
            g1[4] = (int)128u;                            // tile_dim1 = 128 rows
            g1[5] = (int)(unsigned)Kdim;                  // tensor_dim0_stride[31:0]
            g1[6] = 0;
            g1[7] = 0;
            v4i z4; z4[0] = 0; z4[1] = 0; z4[2] = 0; z4[3] = 0;
            v8i z8; for (int zi = 0; zi < 8; ++zi) z8[zi] = 0;
            __builtin_amdgcn_tensor_load_to_lds(g0, g1, z4, z4, z8, 0);
            __builtin_amdgcn_s_wait_tensorcnt(0);
        }
        // --- B tile staged transposed via VGPR path (needs transpose) ---
        for (int li = tid; li < 4096; li += 256) {
            int k = li >> 7, n = li & 127;
            Bt[n * 32 + k] = Bm[(size_t)(k0 + k) * Ndim + blockN + n];
        }
        __syncthreads();

        Frag a[2], bfr[4];
        for (int i = 0; i < 2; ++i) {
            int row = mOff + i * 16 + l16;
            a[i].u[0] = *(const uint4*)&As[row * 32 + kb8];
            a[i].u[1] = *(const uint4*)&As[row * 32 + kb8 + 16];
        }
        for (int i = 0; i < 4; ++i) {
            int n = nOff + i * 16 + l16;
            bfr[i].u[0] = *(const uint4*)&Bt[n * 32 + kb8];
            bfr[i].u[1] = *(const uint4*)&Bt[n * 32 + kb8 + 16];
        }
        for (int mi = 0; mi < 2; ++mi)
            for (int ni = 0; ni < 4; ++ni)
                acc[mi][ni] = wmma_bf16(a[mi], bfr[ni], acc[mi][ni]);
    }

    // Epilogue. C-frag layout: vgpr r, lanes 0-15 -> M=r, lanes 16-31 -> M=r+8
    for (int mi = 0; mi < 2; ++mi) {
        for (int ni = 0; ni < 4; ++ni) {
            for (int r = 0; r < 8; ++r) {
                int row = blockM + mOff + mi * 16 + hi * 8 + r;
                int col = blockN + nOff + ni * 16 + l16;
                float v = acc[mi][ni][r] + bias[col];
                if (mode == 0) {
                    ((float*)out)[(size_t)row * Ndim + col] = v;
                } else if (mode == 1) {
                    ((unsigned short*)out)[(size_t)row * Ndim + col] = f32_to_bf16(v);
                } else if (mode == 2) {
                    ((unsigned short*)out)[(size_t)row * Ndim + col] =
                        f32_to_bf16(gelu_exact(v));
                } else {   // mode 3: final projector, skip memory-token rows
                    int b = row >> 10;
                    int t = row & (TT - 1);
                    if (t >= TM)
                        ((float*)out)[((size_t)b * TS + (t - TM)) * Ndim + col] =
                            gelu_exact(v);
                }
            }
        }
    }
}

// ---------------------------------------------------------------------------
// Flash attention (bf16 WMMA, online softmax). One block per (qblock64, b*h);
// 4 waves, each owns a 16-query x 64-dh output tile. K chunk staged via
// ASYNC global->LDS (ASYNCcnt); V chunk staged transposed via VGPR path.
// ---------------------------------------------------------------------------
__global__ __launch_bounds__(128)
void flash_attn_kernel(const unsigned short* __restrict__ Qb,
                       const unsigned short* __restrict__ Kb,
                       const unsigned short* __restrict__ Vb,
                       unsigned short* __restrict__ Ob) {
    __shared__ __align__(16) unsigned short Ks[32 * 64];      // [key][dh]
    __shared__ __align__(16) unsigned short Vt[64 * 32];      // [dh][key]
    __shared__ __align__(16) unsigned short Ps[4][16 * 32];   // per-wave P tile

    const int tid  = threadIdx.x;
    const int wave = tid >> 5;
    const int lane = tid & 31;
    const int l16  = lane & 15;
    const int hi   = lane >> 4;
    const int kb8  = hi * 8;

    const int bh = blockIdx.y;        // 0..63
    const int b  = bh >> 4;
    const int h  = bh & 15;
    const int q0 = blockIdx.x * 64 + wave * 16;
    const size_t headOff = (size_t)h * TDH;

    Frag qa[2];
    {
        const size_t qrow = ((size_t)b * TT + q0 + l16) * TD + headOff;
        for (int c = 0; c < 2; ++c) {
            qa[c].u[0] = *(const uint4*)(Qb + qrow + c * 32 + kb8);
            qa[c].u[1] = *(const uint4*)(Qb + qrow + c * 32 + kb8 + 16);
        }
    }

    v8f o[4];
    for (int c = 0; c < 4; ++c)
        for (int e = 0; e < 8; ++e) o[c][e] = 0.f;
    float mrow[8], lrow[8];
    for (int r = 0; r < 8; ++r) { mrow[r] = -INFINITY; lrow[r] = 0.f; }

    const float scale = 0.125f;   // 1/sqrt(64)

    for (int kc = 0; kc < TT; kc += 32) {
        __syncthreads();
        // Stage K chunk [32 keys][64 dh] via async global->LDS (16B/lane x2)
        for (int li = tid; li < 256; li += 128) {
            int key = li >> 3, dq = (li & 7) * 8;
            async_load_lds_b128(
                (unsigned)(uintptr_t)&Ks[key * 64 + dq],
                Kb + ((size_t)b * TT + kc + key) * TD + headOff + dq);
        }
        // Stage V chunk transposed: Vt[dh][key]
        for (int li = tid; li < 2048; li += 128) {
            int key = li >> 6, dh = li & 63;
            Vt[dh * 32 + key] = Vb[((size_t)b * TT + kc + key) * TD + headOff + dh];
        }
        wait_asynccnt0();
        __syncthreads();

        // Scores: S[16q x 32keys] via 2 subchunks x 2 K-splits of WMMA
        v8f s[2];
        for (int sc = 0; sc < 2; ++sc) {
            v8f accs;
            for (int e = 0; e < 8; ++e) accs[e] = 0.f;
            int key = sc * 16 + l16;
            for (int c = 0; c < 2; ++c) {
                Frag kf;
                kf.u[0] = *(const uint4*)&Ks[key * 64 + c * 32 + kb8];
                kf.u[1] = *(const uint4*)&Ks[key * 64 + c * 32 + kb8 + 16];
                accs = wmma_bf16(qa[c], kf, accs);
            }
            s[sc] = accs;
        }

        // Online softmax per row (vgpr r -> row hi*8 + r; cols across 16 lanes)
        for (int r = 0; r < 8; ++r) {
            float a0 = s[0][r] * scale;
            float a1 = s[1][r] * scale;
            float mx = fmaxf(a0, a1);
            for (int msk = 1; msk < 16; msk <<= 1)
                mx = fmaxf(mx, __shfl_xor(mx, msk, 32));
            float mnew  = fmaxf(mrow[r], mx);
            float alpha = __expf(mrow[r] - mnew);
            float p0 = __expf(a0 - mnew);
            float p1 = __expf(a1 - mnew);
            float ps = p0 + p1;
            for (int msk = 1; msk < 16; msk <<= 1)
                ps += __shfl_xor(ps, msk, 32);
            lrow[r] = lrow[r] * alpha + ps;
            mrow[r] = mnew;
            for (int c = 0; c < 4; ++c) o[c][r] *= alpha;
            int row = hi * 8 + r;
            Ps[wave][row * 32 + l16]      = f32_to_bf16(p0);
            Ps[wave][row * 32 + 16 + l16] = f32_to_bf16(p1);
        }

        // O += P(16x32) @ V(32x64): A-frag from per-wave LDS, 4 dh B-frags
        Frag pa;
        const unsigned short* pw = Ps[wave];
        pa.u[0] = *(const uint4*)&pw[l16 * 32 + kb8];
        pa.u[1] = *(const uint4*)&pw[l16 * 32 + kb8 + 16];
        for (int c = 0; c < 4; ++c) {
            Frag vf;
            int dh = c * 16 + l16;
            vf.u[0] = *(const uint4*)&Vt[dh * 32 + kb8];
            vf.u[1] = *(const uint4*)&Vt[dh * 32 + kb8 + 16];
            o[c] = wmma_bf16(pa, vf, o[c]);
        }
    }

    for (int r = 0; r < 8; ++r) {
        float inv = 1.0f / lrow[r];
        int row = hi * 8 + r;
        size_t orow = ((size_t)b * TT + q0 + row) * TD + headOff;
        for (int c = 0; c < 4; ++c)
            Ob[orow + c * 16 + l16] = f32_to_bf16(o[c][r] * inv);
    }
}

// ---------------------------------------------------------------------------
// Host-side orchestration
// ---------------------------------------------------------------------------
static inline size_t alignUp(size_t x) { return (x + 255) & ~(size_t)255; }

extern "C" void kernel_launch(void* const* d_in, const int* in_sizes, int n_in,
                              void* d_out, int out_size, void* d_ws, size_t ws_size,
                              hipStream_t stream) {
    (void)in_sizes; (void)n_in; (void)out_size; (void)ws_size;

    const float* hidden = (const float*)d_in[0];
    const float* mem    = (const float*)d_in[1];
    const float* Wq = (const float*)d_in[2];  const float* bq = (const float*)d_in[3];
    const float* Wk = (const float*)d_in[4];  const float* bk = (const float*)d_in[5];
    const float* Wv = (const float*)d_in[6];  const float* bv = (const float*)d_in[7];
    const float* Wo = (const float*)d_in[8];  const float* bo = (const float*)d_in[9];
    const float* ln1g = (const float*)d_in[10]; const float* ln1b = (const float*)d_in[11];
    const float* Wm = (const float*)d_in[12]; const float* bm = (const float*)d_in[13];
    const float* Wr = (const float*)d_in[14]; const float* br = (const float*)d_in[15];
    const float* ln2g = (const float*)d_in[16]; const float* ln2b = (const float*)d_in[17];
    const float* Wp = (const float*)d_in[18]; const float* bp = (const float*)d_in[19];
    float* out = (float*)d_out;

    char* base = (char*)d_ws;
    size_t cur = 0;
    auto take = [&](size_t bytes) { void* p = base + cur; cur = alignUp(cur + bytes); return p; };

    float*          xf    = (float*)take((size_t)NTOK * TD * 4);
    unsigned short* xb    = (unsigned short*)take((size_t)NTOK * TD * 2);
    unsigned short* qb    = (unsigned short*)take((size_t)NTOK * TD * 2);
    unsigned short* kbf   = (unsigned short*)take((size_t)NTOK * TD * 2);
    unsigned short* vbf   = (unsigned short*)take((size_t)NTOK * TD * 2);
    unsigned short* attnb = (unsigned short*)take((size_t)NTOK * TD * 2);
    unsigned short* hb    = (unsigned short*)take((size_t)NTOK * TI * 2);
    float*          c32   = (float*)take((size_t)NTOK * TD * 4);
    unsigned short* w0    = (unsigned short*)take((size_t)TD * TD * 2);
    unsigned short* w1    = (unsigned short*)take((size_t)TD * TD * 2);
    unsigned short* w2    = (unsigned short*)take((size_t)TD * TD * 2);
    unsigned short* w3    = (unsigned short*)take((size_t)TD * TD * 2);
    unsigned short* wbig  = (unsigned short*)take((size_t)TD * TI * 2);

    auto cvt = [&](const float* s, unsigned short* d, int n) {
        cvt_bf16_kernel<<<(n + 255) / 256, 256, 0, stream>>>(s, d, n);
    };
    auto gemm = [&](const unsigned short* A, const unsigned short* Bmat,
                    const float* bias, void* o, int Md, int Kd, int Nd, int mode) {
        dim3 g(Md / 128, Nd / 128);
        gemm_bf16_kernel<<<g, 256, 0, stream>>>(A, Bmat, bias, o, Kd, Nd, mode);
    };

    build_x_kernel<<<(NTOK * TD) / 256, 256, 0, stream>>>(hidden, mem, xf, xb);

    for (int l = 0; l < 2; ++l) {
        const size_t wOff = (size_t)l * TD * TD;
        cvt(Wq + wOff, w0, TD * TD);
        cvt(Wk + wOff, w1, TD * TD);
        cvt(Wv + wOff, w2, TD * TD);
        cvt(Wo + wOff, w3, TD * TD);

        gemm(xb, w0, bq + l * TD, qb,  NTOK, TD, TD, 1);
        gemm(xb, w1, bk + l * TD, kbf, NTOK, TD, TD, 1);
        gemm(xb, w2, bv + l * TD, vbf, NTOK, TD, TD, 1);

        flash_attn_kernel<<<dim3(TT / 64, TB * TH), 128, 0, stream>>>(qb, kbf, vbf, attnb);

        gemm(attnb, w3, bo + l * TD, c32, NTOK, TD, TD, 0);
        resid_ln_kernel<<<NTOK, 256, 0, stream>>>(c32, xf, xb, ln1g + l * TD, ln1b + l * TD);

        cvt(Wm + (size_t)l * TD * TI, wbig, TD * TI);
        gemm(xb, wbig, bm + l * TI, hb, NTOK, TD, TI, 2);

        cvt(Wr + (size_t)l * TI * TD, wbig, TI * TD);
        gemm(hb, wbig, br + l * TD, c32, NTOK, TI, TD, 0);
        resid_ln_kernel<<<NTOK, 256, 0, stream>>>(c32, xf, xb, ln2g + l * TD, ln2b + l * TD);
    }

    cvt(Wp, wbig, TD * TOUT);
    gemm(xb, wbig, bp, out, NTOK, TD, TOUT, 3);

    readout_kernel<<<(TB * TM * TD) / 256, 256, 0, stream>>>(
        xf, out + (size_t)TB * TS * TOUT);
}
